// SEGNO_11098195493444
// MI455X (gfx1250) — compile-verified, matrix operand order checked by reading, not run on hardware
//
#include <hip/hip_runtime.h>
#include <hip/hip_bf16.h>
#include <math.h>

#define NN 10000
#define NE 320000
#define HID 64
#define COORDS_WEIGHT 1.0f
#define EPSN 1e-8f

typedef __bf16 bf16_t;
typedef __attribute__((ext_vector_type(16))) bf16_t v16bf;
typedef __attribute__((ext_vector_type(8)))  bf16_t v8bf;
typedef __attribute__((ext_vector_type(8)))  float  v8f;

#define LDS_FENCE() asm volatile("s_wait_dscnt 0x0" ::: "memory")

__device__ __forceinline__ float silu_f(float x) {
    return x / (1.0f + __expf(-x));
}

// native fp32 atomic add (GLOBAL_ATOMIC_ADD_F32, no CAS loop)
__device__ __forceinline__ void atomic_fadd(float* p, float v) {
    unsafeAtomicAdd(p, v);
}

__device__ __forceinline__ v16bf cat16(v8bf lo, v8bf hi) {
    v16bf r;
#pragma unroll
    for (int i = 0; i < 8; ++i) { r[i] = lo[i]; r[i + 8] = hi[i]; }
    return r;
}

// GEMM of a 16xK LDS A-tile (row-major, row stride `arow` bf16 elems) against a
// pre-packed B matrix (fragments of 32 lanes x 16 bf16, K-mapping baked in).
// Produces 4 N-tiles of 16x16 f32 in C[], using v_wmma_f32_16x16x32_bf16.
template <int KSTEPS>
__device__ __forceinline__ void gemm_tile(const bf16_t* As, int arow,
                                          const bf16_t* __restrict__ Bp,
                                          int lane, v8f C[4]) {
    const int nlo = lane & 15;
    const int hi  = lane >> 4;
#pragma unroll
    for (int t = 0; t < 4; ++t) C[t] = (v8f)(0.0f);
#pragma unroll
    for (int s = 0; s < KSTEPS; ++s) {
        const bf16_t* ap = As + nlo * arow + s * 32 + hi * 8;
        v8bf alo = *(const v8bf*)ap;
        v8bf ahi = *(const v8bf*)(ap + 16);
        v16bf a = cat16(alo, ahi);
#pragma unroll
        for (int t = 0; t < 4; ++t) {
            const bf16_t* bp = Bp + (size_t)(s * 4 + t) * 512 + lane * 16;
            v8bf blo = *(const v8bf*)bp;
            v8bf bhi = *(const v8bf*)(bp + 8);
            v16bf b = cat16(blo, bhi);
            C[t] = __builtin_amdgcn_wmma_f32_16x16x32_bf16(
                false, a, false, b, (short)0, C[t], false, false);
        }
    }
}

// ---------------------------------------------------------------------------
// Pack a K x 64 fp32 weight matrix into bf16 B-fragments (frags = ceil(K/32)*4).
// Element i of lane L covers K = i%8 + 16*(i/8) + 8*(L>=16)  (ISA 16-bit layout).
// ---------------------------------------------------------------------------
__global__ void pack_kernel(const float* __restrict__ W, int K, int frags,
                            bf16_t* __restrict__ dst) {
    int idx = blockIdx.x * blockDim.x + threadIdx.x;
    if (idx >= frags * 512) return;
    int f = idx >> 9;
    int rem = idx & 511;
    int lane = rem >> 4;
    int i = rem & 15;
    int kstep = f >> 2, t = f & 3;
    int nlo = lane & 15, hi = lane >> 4;
    int k = kstep * 32 + (i & 7) + ((i >> 3) << 4) + (hi << 3);
    int n = t * 16 + nlo;
    float v = (k < K) ? W[k * HID + n] : 0.0f;
    dst[(size_t)f * 512 + lane * 16 + i] = (bf16_t)v;
}

__global__ void zero_kernel(float* __restrict__ p, int n) {
    int i = blockIdx.x * blockDim.x + threadIdx.x;
    if (i < n) p[i] = 0.0f;
}

// Embedding: h = his @ emb_w + emb_b; copy loc->x, vel->v.
__global__ void embed_kernel(const float* __restrict__ his,
                             const float* __restrict__ emb_w,
                             const float* __restrict__ emb_b,
                             float* __restrict__ h,
                             const float* __restrict__ loc,
                             const float* __restrict__ vel,
                             float* __restrict__ x, float* __restrict__ v) {
    int idx = blockIdx.x * blockDim.x + threadIdx.x;
    if (idx < NN * HID) {
        int n = idx / HID, f = idx % HID;
        float acc = emb_b[f];
#pragma unroll
        for (int i = 0; i < 4; ++i) acc += his[n * 4 + i] * emb_w[i * HID + f];
        h[idx] = acc;
    }
    if (idx < NN * 3) { x[idx] = loc[idx]; v[idx] = vel[idx]; }
}

// ---------------------------------------------------------------------------
// Edge kernel: one wave per 16-edge tile.  Fused edge-MLP (2 layers) +
// coord-gate MLP, scatter agg / acc / cnt with native fp32 atomics.
// ---------------------------------------------------------------------------
__global__ void __launch_bounds__(256)
edge_kernel(const float* __restrict__ h, const float* __restrict__ x,
            const int* __restrict__ erow, const int* __restrict__ ecol,
            const float* __restrict__ ea,
            const bf16_t* __restrict__ w1p, const bf16_t* __restrict__ w2p,
            const bf16_t* __restrict__ cw1p,
            const float* __restrict__ b1, const float* __restrict__ b2,
            const float* __restrict__ cb1, const float* __restrict__ cw2,
            float* __restrict__ agg, float* __restrict__ accs,
            float* __restrict__ cnt) {
    __shared__ alignas(32) bf16_t sA[8][16][160];
    __shared__ alignas(32) bf16_t sAct[8][16][64];
    __shared__ int   sRow[8][16];
    __shared__ float sCd[8][16][3];
    __shared__ float sGate[8][16];

    const int wv   = threadIdx.x >> 5;
    const int lane = threadIdx.x & 31;
    const int eL   = lane & 15;
    const int hi   = lane >> 4;
    const int nlo  = eL;
    const int tile = blockIdx.x * 8 + wv;   // 2500 blocks * 8 waves == 20000 tiles

    // ---- build 16x160 bf16 A tile: [h_i | h_j | radial | ea | pad] ----
    {
        int e = tile * 16 + eL;
        int r = erow[e], c = ecol[e];
        float d0 = x[r * 3 + 0] - x[c * 3 + 0];
        float d1 = x[r * 3 + 1] - x[c * 3 + 1];
        float d2 = x[r * 3 + 2] - x[c * 3 + 2];
        if (hi == 0) {
            sRow[wv][eL] = r;
            sCd[wv][eL][0] = d0; sCd[wv][eL][1] = d1; sCd[wv][eL][2] = d2;
#pragma unroll
            for (int f = 0; f < 64; ++f) sA[wv][eL][f] = (bf16_t)h[r * HID + f];
#pragma unroll
            for (int f = 0; f < 16; ++f) sA[wv][eL][64 + f] = (bf16_t)h[c * HID + f];
        } else {
#pragma unroll
            for (int f = 16; f < 64; ++f) sA[wv][eL][64 + f] = (bf16_t)h[c * HID + f];
            float radial = d0 * d0 + d1 * d1 + d2 * d2;
            sA[wv][eL][128] = (bf16_t)radial;
            sA[wv][eL][129] = (bf16_t)ea[e * 2 + 0];
            sA[wv][eL][130] = (bf16_t)ea[e * 2 + 1];
#pragma unroll
            for (int f = 131; f < 160; ++f) sA[wv][eL][f] = (bf16_t)0.0f;
        }
    }
    LDS_FENCE();

    v8f C[4];

    // ---- edge MLP layer 1: [16,160]x[160,64], silu ----
    gemm_tile<5>(&sA[wv][0][0], 160, w1p, lane, C);
#pragma unroll
    for (int t = 0; t < 4; ++t) {
        float bias = b1[t * 16 + nlo];
#pragma unroll
        for (int rr = 0; rr < 8; ++rr)
            sAct[wv][rr + 8 * hi][t * 16 + nlo] = (bf16_t)silu_f(C[t][rr] + bias);
    }
    LDS_FENCE();

    // ---- edge MLP layer 2: [16,64]x[64,64], silu -> ef; scatter agg ----
    gemm_tile<2>(&sAct[wv][0][0], 64, w2p, lane, C);
#pragma unroll
    for (int t = 0; t < 4; ++t) {
        float bias = b2[t * 16 + nlo];
#pragma unroll
        for (int rr = 0; rr < 8; ++rr) {
            float v = silu_f(C[t][rr] + bias);
            int m = rr + 8 * hi;
            atomic_fadd(&agg[(size_t)sRow[wv][m] * HID + t * 16 + nlo], v);
            sAct[wv][m][t * 16 + nlo] = (bf16_t)v;
        }
    }
    LDS_FENCE();

    // ---- coord gate: silu(ef@CW1+cb1) . cw2 ----
    gemm_tile<2>(&sAct[wv][0][0], 64, cw1p, lane, C);
    float gsum[8];
#pragma unroll
    for (int rr = 0; rr < 8; ++rr) gsum[rr] = 0.0f;
#pragma unroll
    for (int t = 0; t < 4; ++t) {
        float bias = cb1[t * 16 + nlo];
        float w2v  = cw2[t * 16 + nlo];
#pragma unroll
        for (int rr = 0; rr < 8; ++rr)
            gsum[rr] += silu_f(C[t][rr] + bias) * w2v;
    }
#pragma unroll
    for (int rr = 0; rr < 8; ++rr) {
#pragma unroll
        for (int m = 1; m < 16; m <<= 1)
            gsum[rr] += __shfl_xor(gsum[rr], m, 16);
    }
    if (nlo == 0) {
#pragma unroll
        for (int rr = 0; rr < 8; ++rr) sGate[wv][rr + 8 * hi] = gsum[rr];
    }
    LDS_FENCE();

    if (hi == 0) {
        float g = sGate[wv][eL];
        int r = sRow[wv][eL];
        atomic_fadd(&accs[r * 3 + 0], sCd[wv][eL][0] * g);
        atomic_fadd(&accs[r * 3 + 1], sCd[wv][eL][1] * g);
        atomic_fadd(&accs[r * 3 + 2], sCd[wv][eL][2] * g);
        atomic_fadd(&cnt[r], 1.0f);
    }
}

// ---------------------------------------------------------------------------
// Node kernel: one wave per 16-node tile. node-MLP via WMMA, vel-MLP via WMMA
// + shuffle dot, then fp32 second-order coord/vel update. h_out = h_in + hn.
// ---------------------------------------------------------------------------
__global__ void __launch_bounds__(128)
node_kernel(const float* __restrict__ hin, const float* __restrict__ agg,
            const float* __restrict__ accs, const float* __restrict__ cnt,
            const float* __restrict__ init_vel,
            float* __restrict__ x, float* __restrict__ v,
            float* __restrict__ hout,
            const bf16_t* __restrict__ nw1p, const bf16_t* __restrict__ nw2p,
            const bf16_t* __restrict__ vw1p,
            const float* __restrict__ nb1, const float* __restrict__ nb2,
            const float* __restrict__ vb1, const float* __restrict__ vw2,
            const float* __restrict__ vb2) {
    __shared__ alignas(32) bf16_t sA[4][16][128];
    __shared__ alignas(32) bf16_t sAct[4][16][64];
    __shared__ float sPhi[4][16];

    const int wv   = threadIdx.x >> 5;
    const int lane = threadIdx.x & 31;
    const int eL   = lane & 15;
    const int hi   = lane >> 4;
    const int nlo  = eL;
    const int tile = blockIdx.x * 4 + wv;
    if (tile >= NN / 16) return;   // wave-uniform: EXEC stays all-ones inside

    // ---- build [h | agg] 16x128 tile ----
    {
        int n = tile * 16 + eL;
        const float* src = (hi == 0) ? &hin[(size_t)n * HID] : &agg[(size_t)n * HID];
        int base = (hi == 0) ? 0 : 64;
#pragma unroll
        for (int f = 0; f < 64; ++f) sA[wv][eL][base + f] = (bf16_t)src[f];
    }
    LDS_FENCE();

    v8f C[4];

    // ---- node MLP layer 1: [16,128]x[128,64], silu ----
    gemm_tile<4>(&sA[wv][0][0], 128, nw1p, lane, C);
#pragma unroll
    for (int t = 0; t < 4; ++t) {
        float bias = nb1[t * 16 + nlo];
#pragma unroll
        for (int rr = 0; rr < 8; ++rr)
            sAct[wv][rr + 8 * hi][t * 16 + nlo] = (bf16_t)silu_f(C[t][rr] + bias);
    }
    LDS_FENCE();

    // ---- node MLP layer 2 (no act): hout = hin + hn ----
    gemm_tile<2>(&sAct[wv][0][0], 64, nw2p, lane, C);
#pragma unroll
    for (int t = 0; t < 4; ++t) {
        float bias = nb2[t * 16 + nlo];
#pragma unroll
        for (int rr = 0; rr < 8; ++rr) {
            int n = tile * 16 + rr + 8 * hi;
            int col = t * 16 + nlo;
            hout[(size_t)n * HID + col] = hin[(size_t)n * HID + col] + C[t][rr] + bias;
        }
    }

    // ---- vel MLP: phi = silu(h@VW1+vb1) . vw2 + vb2  (h = cols 0..63 of sA) ----
    gemm_tile<2>(&sA[wv][0][0], 128, vw1p, lane, C);
    float gsum[8];
#pragma unroll
    for (int rr = 0; rr < 8; ++rr) gsum[rr] = 0.0f;
#pragma unroll
    for (int t = 0; t < 4; ++t) {
        float bias = vb1[t * 16 + nlo];
        float w2v  = vw2[t * 16 + nlo];
#pragma unroll
        for (int rr = 0; rr < 8; ++rr)
            gsum[rr] += silu_f(C[t][rr] + bias) * w2v;
    }
#pragma unroll
    for (int rr = 0; rr < 8; ++rr) {
#pragma unroll
        for (int m = 1; m < 16; m <<= 1)
            gsum[rr] += __shfl_xor(gsum[rr], m, 16);
    }
    if (nlo == 0) {
#pragma unroll
        for (int rr = 0; rr < 8; ++rr) sPhi[wv][rr + 8 * hi] = gsum[rr] + vb2[0];
    }
    LDS_FENCE();

    // ---- per-node fp32 coord / velocity update ----
    if (hi == 0) {
        int n = tile * 16 + eL;
        float c = cnt[n]; c = (c > 1.0f) ? c : 1.0f;
        float a0 = accs[n * 3 + 0] / c * COORDS_WEIGHT;
        float a1 = accs[n * 3 + 1] / c * COORDS_WEIGHT;
        float a2 = accs[n * 3 + 2] / c * COORDS_WEIGHT;
        float i0 = init_vel[n * 3 + 0], i1 = init_vel[n * 3 + 1], i2 = init_vel[n * 3 + 2];
        float inv = 1.0f / (sqrtf(i0 * i0 + i1 * i1 + i2 * i2) + EPSN);
        float phi = sPhi[wv][eL];
        float v0 = v[n * 3 + 0] + a0 + phi * i0 * inv;
        float v1 = v[n * 3 + 1] + a1 + phi * i1 * inv;
        float v2 = v[n * 3 + 2] + a2 + phi * i2 * inv;
        v[n * 3 + 0] = v0; v[n * 3 + 1] = v1; v[n * 3 + 2] = v2;
        x[n * 3 + 0] += v0; x[n * 3 + 1] += v1; x[n * 3 + 2] += v2;
    }
}

// d_out = [x(30000) | h(640000) | v(30000)]
__global__ void finalize_kernel(const float* __restrict__ x,
                                const float* __restrict__ h,
                                const float* __restrict__ v,
                                float* __restrict__ out) {
    int i = blockIdx.x * blockDim.x + threadIdx.x;
    if (i < NN * 3)                 out[i] = x[i];
    else if (i < NN * 3 + NN * HID) out[i] = h[i - NN * 3];
    else if (i < NN * 6 + NN * HID) out[i] = v[i - NN * 3 - NN * HID];
}

extern "C" void kernel_launch(void* const* d_in, const int* in_sizes, int n_in,
                              void* d_out, int out_size, void* d_ws, size_t ws_size,
                              hipStream_t stream) {
    const float* his      = (const float*)d_in[0];
    const float* loc      = (const float*)d_in[1];
    const int*   edges    = (const int*)  d_in[2];
    const float* vel      = (const float*)d_in[3];
    const float* ea       = (const float*)d_in[4];
    const float* emb_w    = (const float*)d_in[5];
    const float* emb_b    = (const float*)d_in[6];
    const float* edge_w1  = (const float*)d_in[7];
    const float* edge_b1  = (const float*)d_in[8];
    const float* edge_w2  = (const float*)d_in[9];
    const float* edge_b2  = (const float*)d_in[10];
    const float* node_w1  = (const float*)d_in[11];
    const float* node_b1  = (const float*)d_in[12];
    const float* node_w2  = (const float*)d_in[13];
    const float* node_b2  = (const float*)d_in[14];
    const float* coord_w1 = (const float*)d_in[15];
    const float* coord_b1 = (const float*)d_in[16];
    const float* coord_w2 = (const float*)d_in[17];
    const float* vel_w1   = (const float*)d_in[18];
    const float* vel_b1   = (const float*)d_in[19];
    const float* vel_w2   = (const float*)d_in[20];
    const float* vel_b2   = (const float*)d_in[21];

    const int* erow = edges;
    const int* ecol = edges + NE;

    float* ws = (float*)d_ws;
    float* hA   = ws;                 // NN*HID
    float* hB   = hA + NN * HID;
    float* xb   = hB + NN * HID;      // NN*3
    float* vb   = xb + NN * 3;        // NN*3
    float* agg  = vb + NN * 3;        // NN*HID
    float* accs = agg + NN * HID;     // NN*3
    float* cntb = accs + NN * 3;      // NN
    bf16_t* w1p  = (bf16_t*)(cntb + NN);   // 20*512
    bf16_t* w2p  = w1p  + 20 * 512;        // 8*512
    bf16_t* cw1p = w2p  + 8 * 512;
    bf16_t* nw1p = cw1p + 8 * 512;         // 16*512
    bf16_t* nw2p = nw1p + 16 * 512;
    bf16_t* vw1p = nw2p + 8 * 512;

    // pack weights to bf16 B-fragments
    pack_kernel<<<(20 * 512 + 255) / 256, 256, 0, stream>>>(edge_w1, 131, 20, w1p);
    pack_kernel<<<(8  * 512 + 255) / 256, 256, 0, stream>>>(edge_w2,  64,  8, w2p);
    pack_kernel<<<(8  * 512 + 255) / 256, 256, 0, stream>>>(coord_w1, 64,  8, cw1p);
    pack_kernel<<<(16 * 512 + 255) / 256, 256, 0, stream>>>(node_w1, 128, 16, nw1p);
    pack_kernel<<<(8  * 512 + 255) / 256, 256, 0, stream>>>(node_w2,  64,  8, nw2p);
    pack_kernel<<<(8  * 512 + 255) / 256, 256, 0, stream>>>(vel_w1,   64,  8, vw1p);

    // embedding + state init
    embed_kernel<<<(NN * HID + 255) / 256, 256, 0, stream>>>(
        his, emb_w, emb_b, hA, loc, vel, xb, vb);

    const int zcount = NN * HID + NN * 3 + NN;   // agg, accs, cnt contiguous
    for (int layer = 0; layer < 4; ++layer) {
        const float* hin = (layer & 1) ? hB : hA;
        float* houtp     = (layer & 1) ? hA : hB;
        zero_kernel<<<(zcount + 255) / 256, 256, 0, stream>>>(agg, zcount);
        edge_kernel<<<NE / 16 / 8, 256, 0, stream>>>(
            hin, xb, erow, ecol, ea, w1p, w2p, cw1p,
            edge_b1, edge_b2, coord_b1, coord_w2, agg, accs, cntb);
        node_kernel<<<(NN / 16 + 3) / 4, 128, 0, stream>>>(
            hin, agg, accs, cntb, vel, xb, vb, houtp,
            nw1p, nw2p, vw1p, node_b1, node_b2, vel_b1, vel_w2, vel_b2);
    }
    // after 4 layers (0:A->B, 1:B->A, 2:A->B, 3:B->A) final h is in hA
    finalize_kernel<<<(NN * 6 + NN * HID + 255) / 256, 256, 0, stream>>>(
        xb, hA, vb, (float*)d_out);
}
